// GATmodel_10428180595388
// MI455X (gfx1250) — matrix operand order repaired and checked by reading, not compiled, not run on hardware
//
#include <hip/hip_runtime.h>
#include <cmath>

// ---------------- sizes from the reference ----------------
#define IN_F   128
#define HID    64
#define HEADS  3
#define OUTF   (HEADS * HID)   // 192
#define NEG_SLOPE 0.2f

typedef __attribute__((ext_vector_type(16))) _Float16 v16h;
typedef __attribute__((ext_vector_type(8)))  float    v8f;

union Frag16 { v16h v; uint4 u[2]; };

// ---- WMMA GEMM: C[M,Nout] = A[M,K](f16) * Bt[Nout,K](f16)^T --------------
// One wave computes a 16x64 output strip: one A-frag feeds 4 B-frags/accums,
// cutting A traffic 4x vs a 16x16 tile (GEMM is fetch-limited, WMMA is fast).
// A-frag (16x32 f16): lane<16 row=lane, K {k..k+7, k+16..k+23};
//                     lane>=16 row=lane-16, K {k+8..k+15, k+24..k+31}.
// B-frag (32x16 f16): lane<16 col=lane, K {k..k+7, k+16..k+23}; lane>=16 col-16, K {k+8..k+15, k+24..k+31}.
// D: lane<16 -> rows 0..7, lane>=16 -> rows 8..15, col = lane&15.
__global__ __launch_bounds__(32)
void wmma_gemm_f16(const _Float16* __restrict__ A,
                   const _Float16* __restrict__ Bt,
                   float* __restrict__ C,
                   int K, int Nout)
{
    const int lane  = threadIdx.x;
    const int l16   = lane & 15;
    const bool hi   = lane >= 16;
    const int nBase = blockIdx.x * 64;
    const int mBase = blockIdx.y * 16;

    const _Float16* arow = A  + (size_t)(mBase + l16) * K;
    const _Float16* brow = Bt + (size_t)(nBase + l16) * K;

    v8f acc0 = {}, acc1 = {}, acc2 = {}, acc3 = {};
    for (int k = 0; k < K; k += 32) {
        Frag16 a, b0, b1, b2, b3;
        const _Float16* ap = arow + k + (hi ? 8 : 0);
        a.u[0] = *(const uint4*)(ap);        // 8 halves
        a.u[1] = *(const uint4*)(ap + 16);   // +16 halves = +32B

        const _Float16* bp = brow + k + (hi ? 8 : 0);
        b0.u[0] = *(const uint4*)(bp);
        b0.u[1] = *(const uint4*)(bp + 16);
        b1.u[0] = *(const uint4*)(bp + (size_t)16 * K);
        b1.u[1] = *(const uint4*)(bp + (size_t)16 * K + 16);
        b2.u[0] = *(const uint4*)(bp + (size_t)32 * K);
        b2.u[1] = *(const uint4*)(bp + (size_t)32 * K + 16);
        b3.u[0] = *(const uint4*)(bp + (size_t)48 * K);
        b3.u[1] = *(const uint4*)(bp + (size_t)48 * K + 16);

        acc0 = __builtin_amdgcn_wmma_f32_16x16x32_f16(false, a.v, false, b0.v, (short)0, acc0, false, false);
        acc1 = __builtin_amdgcn_wmma_f32_16x16x32_f16(false, a.v, false, b1.v, (short)0, acc1, false, false);
        acc2 = __builtin_amdgcn_wmma_f32_16x16x32_f16(false, a.v, false, b2.v, (short)0, acc2, false, false);
        acc3 = __builtin_amdgcn_wmma_f32_16x16x32_f16(false, a.v, false, b3.v, (short)0, acc3, false, false);
    }

    float* crow = C + (size_t)(mBase + (hi ? 8 : 0)) * Nout + (nBase + l16);
#pragma unroll
    for (int i = 0; i < 8; ++i) {
        crow[(size_t)i * Nout]      = acc0[i];
        crow[(size_t)i * Nout + 16] = acc1[i];
        crow[(size_t)i * Nout + 32] = acc2[i];
        crow[(size_t)i * Nout + 48] = acc3[i];
    }
}

// ---------------- helpers ----------------
__global__ void f32_to_f16_kernel(const float* __restrict__ in,
                                  _Float16* __restrict__ out, int n)
{
    int i = blockIdx.x * blockDim.x + threadIdx.x;
    if (i < n) out[i] = (_Float16)in[i];
}

// W[K, Nout] f32 -> Wt[Nout, K] f16
__global__ void transpose_to_f16_kernel(const float* __restrict__ W,
                                        _Float16* __restrict__ Wt,
                                        int K, int Nout)
{
    int i = blockIdx.x * blockDim.x + threadIdx.x;
    if (i >= K * Nout) return;
    int k = i / Nout, nn = i % Nout;
    Wt[(size_t)nn * K + k] = (_Float16)W[i];
}

__global__ void fill_kernel(float* __restrict__ p, float v, int n)
{
    int i = blockIdx.x * blockDim.x + threadIdx.x;
    if (i < n) p[i] = v;
}

// alpha_src[n,h] = dot(h[n,h,:], a_src[h,:]) ; same for dst
__global__ void alpha_kernel(const float* __restrict__ h,
                             const float* __restrict__ a_src,
                             const float* __restrict__ a_dst,
                             float* __restrict__ asrc,
                             float* __restrict__ adst, int N)
{
    int i = blockIdx.x * blockDim.x + threadIdx.x;   // n*HEADS + head
    if (i >= N * HEADS) return;
    int n = i / HEADS, hh = i % HEADS;
    const float* hp = h + (size_t)n * OUTF + hh * HID;
    const float* as = a_src + hh * HID;
    const float* ad = a_dst + hh * HID;
    float s0 = 0.f, s1 = 0.f;
#pragma unroll 4
    for (int c = 0; c < HID; ++c) {
        float hv = hp[c];
        s0 += hv * as[c];
        s1 += hv * ad[c];
    }
    asrc[i] = s0;
    adst[i] = s1;
}

__device__ __forceinline__ float lrelu(float x)
{
    return x > 0.f ? x : NEG_SLOPE * x;
}

__device__ __forceinline__ void atomicMaxF(float* a, float v)
{
    if (v >= 0.f) atomicMax((int*)a, __float_as_int(v));
    else          atomicMin((unsigned int*)a, __float_as_uint(v));
}

// pass 1: segment max over edges (self-loops appended as i>=E)
__global__ void edge_max_kernel(const int* __restrict__ src,
                                const int* __restrict__ dst,
                                const float* __restrict__ asrc,
                                const float* __restrict__ adst,
                                float* __restrict__ m, int E, int N)
{
    int i = blockIdx.x * blockDim.x + threadIdx.x;
    if (i >= E + N) return;
    int s = (i < E) ? src[i] : (i - E);
    int d = (i < E) ? dst[i] : (i - E);
#pragma unroll
    for (int hh = 0; hh < HEADS; ++hh) {
        float e = lrelu(asrc[s * HEADS + hh] + adst[d * HEADS + hh]);
        atomicMaxF(&m[d * HEADS + hh], e);
    }
}

// pass 2: segment sum of exp(e - m[d])
__global__ void edge_sum_kernel(const int* __restrict__ src,
                                const int* __restrict__ dst,
                                const float* __restrict__ asrc,
                                const float* __restrict__ adst,
                                const float* __restrict__ m,
                                float* __restrict__ denom, int E, int N)
{
    int i = blockIdx.x * blockDim.x + threadIdx.x;
    if (i >= E + N) return;
    int s = (i < E) ? src[i] : (i - E);
    int d = (i < E) ? dst[i] : (i - E);
#pragma unroll
    for (int hh = 0; hh < HEADS; ++hh) {
        float e = lrelu(asrc[s * HEADS + hh] + adst[d * HEADS + hh]);
        atomicAdd(&denom[d * HEADS + hh], expf(e - m[d * HEADS + hh]));
    }
}

// pass 3: acc[d, c] += (1/3) * sum_h alpha[e,h] * hfeat[s, h, c]
// 16 lanes cooperate per edge: lane j covers channels 4j..4j+3 with one
// float4 load per head -> coalesced 256B segments, 4 atomics per lane.
__global__ void edge_agg_kernel(const int* __restrict__ src,
                                const int* __restrict__ dst,
                                const float* __restrict__ asrc,
                                const float* __restrict__ adst,
                                const float* __restrict__ m,
                                const float* __restrict__ denom,
                                const float* __restrict__ hfeat,
                                float* __restrict__ acc, int E, int N)
{
    int t = blockIdx.x * blockDim.x + threadIdx.x;
    int i = t >> 4;          // edge index
    int j = t & 15;          // channel group within edge
    if (i >= E + N) return;
    int s = (i < E) ? src[i] : (i - E);
    int d = (i < E) ? dst[i] : (i - E);

    float al[HEADS];
#pragma unroll
    for (int hh = 0; hh < HEADS; ++hh) {
        float e = lrelu(asrc[s * HEADS + hh] + adst[d * HEADS + hh]);
        float p = expf(e - m[d * HEADS + hh]);
        al[hh] = p / (denom[d * HEADS + hh] + 1e-16f) * (1.f / HEADS);
    }

    const float* hp = hfeat + (size_t)s * OUTF + 4 * j;
    float4 v0 = *(const float4*)(hp);
    float4 v1 = *(const float4*)(hp + HID);
    float4 v2 = *(const float4*)(hp + 2 * HID);

    float* ad = acc + (size_t)d * HID + 4 * j;
    atomicAdd(&ad[0], al[0] * v0.x + al[1] * v1.x + al[2] * v2.x);
    atomicAdd(&ad[1], al[0] * v0.y + al[1] * v1.y + al[2] * v2.y);
    atomicAdd(&ad[2], al[0] * v0.z + al[1] * v1.z + al[2] * v2.z);
    atomicAdd(&ad[3], al[0] * v0.w + al[1] * v1.w + al[2] * v2.w);
}

// layer-1 epilogue: relu(acc + bias) -> f16 activations for layer-2 GEMM
__global__ void bias_relu_f16_kernel(const float* __restrict__ acc,
                                     const float* __restrict__ bias,
                                     _Float16* __restrict__ out16, int N)
{
    int i = blockIdx.x * blockDim.x + threadIdx.x;
    if (i >= N * HID) return;
    float v = acc[i] + bias[i & (HID - 1)];
    out16[i] = (_Float16)(v > 0.f ? v : 0.f);
}

// layer-2 epilogue: acc + bias -> f32 output
__global__ void bias_out_kernel(const float* __restrict__ acc,
                                const float* __restrict__ bias,
                                float* __restrict__ out, int N)
{
    int i = blockIdx.x * blockDim.x + threadIdx.x;
    if (i >= N * HID) return;
    out[i] = acc[i] + bias[i & (HID - 1)];
}

// ---------------- host-side orchestration ----------------
extern "C" void kernel_launch(void* const* d_in, const int* in_sizes, int n_in,
                              void* d_out, int out_size, void* d_ws, size_t ws_size,
                              hipStream_t stream)
{
    const float* x        = (const float*)d_in[0];
    const int*   ei       = (const int*)  d_in[1];
    const float* W1       = (const float*)d_in[2];
    const float* att_src1 = (const float*)d_in[3];
    const float* att_dst1 = (const float*)d_in[4];
    const float* bias1    = (const float*)d_in[5];
    const float* W2       = (const float*)d_in[6];
    const float* att_src2 = (const float*)d_in[7];
    const float* att_dst2 = (const float*)d_in[8];
    const float* bias2    = (const float*)d_in[9];

    const int N = in_sizes[0] / IN_F;       // 50000
    const int E = in_sizes[1] / 2;          // 800000
    const int* src = ei;
    const int* dst = ei + E;

    // ---- carve workspace (256B aligned) ----
    char* p = (char*)d_ws;
    auto carve = [&](size_t bytes) -> char* {
        char* r = p;
        p += (bytes + 255) & ~(size_t)255;
        return r;
    };
    float*    hbuf  = (float*)   carve((size_t)N * OUTF * sizeof(float));     // 38.4 MB
    _Float16* af16  = (_Float16*)carve((size_t)N * IN_F * sizeof(_Float16));  // 12.8 MB
    _Float16* wt16  = (_Float16*)carve((size_t)OUTF * IN_F * sizeof(_Float16));
    float*    asrc  = (float*)   carve((size_t)N * HEADS * sizeof(float));
    float*    adst  = (float*)   carve((size_t)N * HEADS * sizeof(float));
    float*    mbuf  = (float*)   carve((size_t)N * HEADS * sizeof(float));
    float*    dbuf  = (float*)   carve((size_t)N * HEADS * sizeof(float));
    float*    acc   = (float*)   carve((size_t)N * HID * sizeof(float));      // 12.8 MB
    (void)ws_size;

    const int T = 256;
    const float NEG_INF = -INFINITY;
    dim3 gemmGrid(OUTF / 64, N / 16);   // 3 x 3125 waves, 16x64 strip each
    int edgeBlocks  = (E + N + T - 1) / T;
    int aggBlocks   = ((E + N) * 16 + T - 1) / T;
    int nodeHBlk    = (N * HEADS + T - 1) / T;
    int nodeCBlk    = (N * HID + T - 1) / T;

    // ================= layer 1 =================
    f32_to_f16_kernel<<<(N * IN_F + T - 1) / T, T, 0, stream>>>(x, af16, N * IN_F);
    transpose_to_f16_kernel<<<(IN_F * OUTF + T - 1) / T, T, 0, stream>>>(W1, wt16, IN_F, OUTF);
    wmma_gemm_f16<<<gemmGrid, 32, 0, stream>>>(af16, wt16, hbuf, IN_F, OUTF);
    alpha_kernel<<<nodeHBlk, T, 0, stream>>>(hbuf, att_src1, att_dst1, asrc, adst, N);
    fill_kernel<<<nodeHBlk, T, 0, stream>>>(mbuf, NEG_INF, N * HEADS);
    fill_kernel<<<nodeHBlk, T, 0, stream>>>(dbuf, 0.f, N * HEADS);
    fill_kernel<<<nodeCBlk, T, 0, stream>>>(acc, 0.f, N * HID);
    edge_max_kernel<<<edgeBlocks, T, 0, stream>>>(src, dst, asrc, adst, mbuf, E, N);
    edge_sum_kernel<<<edgeBlocks, T, 0, stream>>>(src, dst, asrc, adst, mbuf, dbuf, E, N);
    edge_agg_kernel<<<aggBlocks, T, 0, stream>>>(src, dst, asrc, adst, mbuf, dbuf, hbuf, acc, E, N);
    bias_relu_f16_kernel<<<nodeCBlk, T, 0, stream>>>(acc, bias1, af16, N);   // af16 now [N,64] f16

    // ================= layer 2 =================
    transpose_to_f16_kernel<<<(HID * OUTF + T - 1) / T, T, 0, stream>>>(W2, wt16, HID, OUTF);
    wmma_gemm_f16<<<gemmGrid, 32, 0, stream>>>(af16, wt16, hbuf, HID, OUTF);
    alpha_kernel<<<nodeHBlk, T, 0, stream>>>(hbuf, att_src2, att_dst2, asrc, adst, N);
    fill_kernel<<<nodeHBlk, T, 0, stream>>>(mbuf, NEG_INF, N * HEADS);
    fill_kernel<<<nodeHBlk, T, 0, stream>>>(dbuf, 0.f, N * HEADS);
    fill_kernel<<<nodeCBlk, T, 0, stream>>>(acc, 0.f, N * HID);
    edge_max_kernel<<<edgeBlocks, T, 0, stream>>>(src, dst, asrc, adst, mbuf, E, N);
    edge_sum_kernel<<<edgeBlocks, T, 0, stream>>>(src, dst, asrc, adst, mbuf, dbuf, E, N);
    edge_agg_kernel<<<aggBlocks, T, 0, stream>>>(src, dst, asrc, adst, mbuf, dbuf, hbuf, acc, E, N);
    bias_out_kernel<<<nodeCBlk, T, 0, stream>>>(acc, bias2, (float*)d_out, N);
}
